// MultiScaleRetentionTriton_55009941127828
// MI455X (gfx1250) — matrix operand order repaired and checked by reading, not compiled, not run on hardware
//
#include <hip/hip_runtime.h>
#include <hip/hip_bf16.h>
#include <stdint.h>

// ---------------------------------------------------------------------------
// MultiScaleRetention forward for MI455X (gfx1250), bf16 WMMA pipeline.
//   q = x@Wq.T ; v = x@Wv.T ; g = x@Wg.T          (WMMA bf16 -> f32)
//   state_t = lam[h]*state_{t-1} + v_t ; y = q*state   (per-channel scan)
//   y = LN(y)*silu(g)                               (row reduce + elementwise)
//   out = y@Wo.T                                    (WMMA bf16 -> f32)
// GEMM inner loop is register double-buffered and pinned with
// sched_barrier(0) so next-step loads stay in flight across current WMMAs.
// ---------------------------------------------------------------------------

typedef __attribute__((ext_vector_type(16))) __bf16 v16bf;
typedef __attribute__((ext_vector_type(8)))  float  v8f;

#define D_MODEL  1024
#define NHEAD    16
#define HEAD_DIM 64
#define BATCH    4
#define SEQ      4096
#define MROWS    (BATCH * SEQ)     // 16384
#define LN_EPS   1e-5f

#if defined(__has_builtin)
#if __has_builtin(__builtin_amdgcn_sched_barrier)
#define SCHED_FENCE() __builtin_amdgcn_sched_barrier(0)
#endif
#endif
#ifndef SCHED_FENCE
#define SCHED_FENCE()
#endif

union ABfrag { v16bf v; uint4 u4[2]; };
union Pack4  { uint2 u; __bf16 b[4]; };

// ---------------------------- fp32 -> bf16 convert -------------------------
__global__ __launch_bounds__(256)
void cvt_f32_bf16(const float* __restrict__ in, __bf16* __restrict__ out, int n4) {
    int i = blockIdx.x * blockDim.x + threadIdx.x;
    if (i >= n4) return;
    float4 f = ((const float4*)in)[i];
    Pack4 p;
    p.b[0] = (__bf16)f.x; p.b[1] = (__bf16)f.y;
    p.b[2] = (__bf16)f.z; p.b[3] = (__bf16)f.w;
    ((uint2*)out)[i] = p.u;
}

// ---------------------- GEMM: C[M,N] = A[M,K] @ W[N,K]^T -------------------
// A row-major bf16, W row-major bf16 (reference x @ W.T => B[k][n] = W[n][k],
// which is K-contiguous in W -> plain b128 loads, no transpose needed).
// Wave tile: 16 (M) x 64 (N), 4 accumulators sharing one A fragment.
// Block: 8 waves stacked in M -> 128 x 64 block tile.
template <bool BF16OUT, int N, int K>
__global__ __launch_bounds__(256)
void gemm_xwT(const __bf16* __restrict__ A, const __bf16* __restrict__ W,
              void* __restrict__ C) {
    const int lane = threadIdx.x & 31;
    const int wave = threadIdx.x >> 5;        // 0..7
    const int half = lane >> 4;               // lane group (0:lanes0-15, 1:16-31)
    const int mr   = lane & 15;
    const int row0 = (blockIdx.y * 8 + wave) * 16;
    const int col0 = blockIdx.x * 64;

    // A fragment (16x32 bf16): VGPR0-3 = K[h*8 .. h*8+7],
    //                          VGPR4-7 = K[16+h*8 .. 16+h*8+7]
    const __bf16* arow = A + (size_t)(row0 + mr) * K + half * 8;
    // B fragment (32x16 bf16): lane holds column n, K = half*8 .. half*8+7
    // and 16+half*8 .. 16+half*8+7 of W row n (same order as A).
    const __bf16* wbase[4];
#pragma unroll
    for (int t = 0; t < 4; ++t)
        wbase[t] = W + (size_t)(col0 + t * 16 + mr) * K + half * 8;

    ABfrag a_cur, a_nxt, b_cur[4], b_nxt[4];

    // prologue: k = 0
    a_cur.u4[0] = *(const uint4*)(arow);
    a_cur.u4[1] = *(const uint4*)(arow + 16);
#pragma unroll
    for (int t = 0; t < 4; ++t) {
        b_cur[t].u4[0] = ((const uint4*)wbase[t])[0];
        b_cur[t].u4[1] = ((const uint4*)wbase[t])[2];
    }

    v8f acc[4] = {};
    for (int k = 32; k < K; k += 32) {
        // issue next step's loads before this step's WMMAs (double buffer)
        a_nxt.u4[0] = *(const uint4*)(arow + k);
        a_nxt.u4[1] = *(const uint4*)(arow + k + 16);
#pragma unroll
        for (int t = 0; t < 4; ++t) {
            b_nxt[t].u4[0] = ((const uint4*)(wbase[t] + k))[0];
            b_nxt[t].u4[1] = ((const uint4*)(wbase[t] + k))[2];
        }
        __builtin_prefetch(arow + k + 256, 0, 1);   // global_prefetch_b8 ahead in K
        // Hard scheduling fence: WMMAs below may not hoist above the loads,
        // loads above may not sink below -> both buffers stay live, waits
        // become partial loadcnt thresholds instead of wait-to-zero.
        SCHED_FENCE();
#pragma unroll
        for (int t = 0; t < 4; ++t)
            acc[t] = __builtin_amdgcn_wmma_f32_16x16x32_bf16(
                false, a_cur.v, false, b_cur[t].v, (short)0, acc[t], false, false);
        SCHED_FENCE();
        a_cur = a_nxt;
#pragma unroll
        for (int t = 0; t < 4; ++t) b_cur[t] = b_nxt[t];
    }
    // epilogue: last k-step
#pragma unroll
    for (int t = 0; t < 4; ++t)
        acc[t] = __builtin_amdgcn_wmma_f32_16x16x32_bf16(
            false, a_cur.v, false, b_cur[t].v, (short)0, acc[t], false, false);

    // C/D layout: VGPR r -> (M = row0 + r + 8*half, N = col0 + t*16 + mr)
    const int crow0 = row0 + half * 8;
    if constexpr (BF16OUT) {
        __bf16* Cp = (__bf16*)C;
#pragma unroll
        for (int t = 0; t < 4; ++t)
#pragma unroll
            for (int r = 0; r < 8; ++r)
                Cp[(size_t)(crow0 + r) * N + col0 + t * 16 + mr] = (__bf16)acc[t][r];
    } else {
        float* Cp = (float*)C;
#pragma unroll
        for (int t = 0; t < 4; ++t)
#pragma unroll
            for (int r = 0; r < 8; ++r)
                Cp[(size_t)(crow0 + r) * N + col0 + t * 16 + mr] = acc[t][r];
    }
}

// ------------------------- retention scan (per channel) --------------------
// One thread per (b, h, dh): state = lam[h]*state + v_t ; y = q_t * state.
// 4096 threads = 128 waves; fp32 state in registers, bf16 streaming I/O.
__global__ __launch_bounds__(256)
void retention_scan(const __bf16* __restrict__ q, const __bf16* __restrict__ v,
                    const float* __restrict__ decay, __bf16* __restrict__ y) {
    const int idx = blockIdx.x * blockDim.x + threadIdx.x;   // < BATCH*NHEAD*HEAD_DIM
    const int dh = idx % HEAD_DIM;
    const int h  = (idx / HEAD_DIM) % NHEAD;
    const int b  = idx / (HEAD_DIM * NHEAD);
    const float lam = decay[h * 2 + 1];                      // decay[h,1,0]
    float state = 0.f;
    size_t off = (size_t)b * SEQ * D_MODEL + (size_t)h * HEAD_DIM + dh;
    for (int l = 0; l < SEQ; ++l) {
        state = fmaf(lam, state, (float)v[off]);
        y[off] = (__bf16)((float)q[off] * state);
        off += D_MODEL;
    }
}

// --------------------- LayerNorm + SiLU gate (per row) ---------------------
__global__ __launch_bounds__(256)
void ln_silu_gate(const __bf16* __restrict__ y, const __bf16* __restrict__ g,
                  const float* __restrict__ ln_g, const float* __restrict__ ln_b,
                  __bf16* __restrict__ out) {
    __shared__ float redS[8], redQ[8];
    const int row  = blockIdx.x;
    const int t    = threadIdx.x;
    const int lane = t & 31;
    const int wave = t >> 5;

    const __bf16* yr = y + (size_t)row * D_MODEL;
    const __bf16* gr = g + (size_t)row * D_MODEL;

    Pack4 yp, gp;
    yp.u = ((const uint2*)yr)[t];
    gp.u = ((const uint2*)gr)[t];

    float f[4], s = 0.f, ss = 0.f;
#pragma unroll
    for (int j = 0; j < 4; ++j) {
        f[j] = (float)yp.b[j];
        s += f[j];
        ss = fmaf(f[j], f[j], ss);
    }
    // wave32 reduction
#pragma unroll
    for (int off = 16; off >= 1; off >>= 1) {
        s  += __shfl_xor(s, off, 32);
        ss += __shfl_xor(ss, off, 32);
    }
    if (lane == 0) { redS[wave] = s; redQ[wave] = ss; }
    __syncthreads();
    s = 0.f; ss = 0.f;
#pragma unroll
    for (int w = 0; w < 8; ++w) { s += redS[w]; ss += redQ[w]; }

    const float inv = 1.f / (float)D_MODEL;
    const float mu  = s * inv;
    const float var = ss * inv - mu * mu;
    const float rs  = rsqrtf(var + LN_EPS);

    Pack4 op;
#pragma unroll
    for (int j = 0; j < 4; ++j) {
        const int c   = t * 4 + j;
        const float yv = ((f[j] - mu) * rs) * ln_g[c] + ln_b[c];
        const float gv = (float)gp.b[j];
        const float gate = gv * (1.f / (1.f + __expf(-gv)));   // silu
        op.b[j] = (__bf16)(yv * gate);
    }
    ((uint2*)(out + (size_t)row * D_MODEL))[t] = op.u;
}

// ---------------------------------------------------------------------------
extern "C" void kernel_launch(void* const* d_in, const int* in_sizes, int n_in,
                              void* d_out, int out_size, void* d_ws, size_t ws_size,
                              hipStream_t stream) {
    (void)in_sizes; (void)n_in; (void)out_size; (void)ws_size;
    const float* x     = (const float*)d_in[0];
    const float* decay = (const float*)d_in[3];
    const float* Wq    = (const float*)d_in[4];
    const float* Wv    = (const float*)d_in[5];
    const float* Wo    = (const float*)d_in[6];
    const float* Wg    = (const float*)d_in[7];
    const float* ln_g  = (const float*)d_in[8];
    const float* ln_b  = (const float*)d_in[9];

    const size_t NX = (size_t)MROWS * D_MODEL;     // 16.78M activations
    const size_t NW = (size_t)D_MODEL * D_MODEL;   // 1.05M per weight

    __bf16* xbf = (__bf16*)d_ws;       // 32 MB
    __bf16* wqb = xbf + NX;            //  2 MB
    __bf16* wvb = wqb + NW;
    __bf16* wgb = wvb + NW;
    __bf16* wob = wgb + NW;
    __bf16* qb  = wob + NW;            // 32 MB each below
    __bf16* vb  = qb + NX;
    __bf16* gb  = vb + NX;
    __bf16* yb  = gb + NX;
    __bf16* zb  = yb + NX;

    // Stage 0: downconvert activations + weights to bf16 (L2-resident after).
    cvt_f32_bf16<<<(int)(NX / 4 / 256), 256, 0, stream>>>(x,  xbf, (int)(NX / 4));
    cvt_f32_bf16<<<(int)(NW / 4 / 256), 256, 0, stream>>>(Wq, wqb, (int)(NW / 4));
    cvt_f32_bf16<<<(int)(NW / 4 / 256), 256, 0, stream>>>(Wv, wvb, (int)(NW / 4));
    cvt_f32_bf16<<<(int)(NW / 4 / 256), 256, 0, stream>>>(Wg, wgb, (int)(NW / 4));
    cvt_f32_bf16<<<(int)(NW / 4 / 256), 256, 0, stream>>>(Wo, wob, (int)(NW / 4));

    // Stage 1: q, v, pre-gate GEMMs (WMMA bf16).
    dim3 gg(D_MODEL / 64, MROWS / 128);
    gemm_xwT<true, D_MODEL, D_MODEL><<<gg, 256, 0, stream>>>(xbf, wqb, qb);
    gemm_xwT<true, D_MODEL, D_MODEL><<<gg, 256, 0, stream>>>(xbf, wvb, vb);
    gemm_xwT<true, D_MODEL, D_MODEL><<<gg, 256, 0, stream>>>(xbf, wgb, gb);

    // Stage 2: decay scan along L.
    retention_scan<<<(BATCH * NHEAD * HEAD_DIM) / 256, 256, 0, stream>>>(qb, vb, decay, yb);

    // Stage 3: LayerNorm + SiLU gate.
    ln_silu_gate<<<MROWS, 256, 0, stream>>>(yb, gb, ln_g, ln_b, zb);

    // Stage 4: output projection (fp32 out).
    gemm_xwT<false, D_MODEL, D_MODEL><<<gg, 256, 0, stream>>>(zb, wob, d_out);
}
